// DSPP_14345190768897
// MI455X (gfx1250) — compile-verified, matrix operand order checked by reading, not run on hardware
//
#include <hip/hip_runtime.h>

// ---------------- problem constants ----------------
#define B_    2
#define CIN   2048
#define HW    1024      // 32*32
#define M_TOT 2048      // B_*HW  (GEMM M dimension, pixels across batch)
#define COUT  256
#define CCAT  1280

// ---------------- CDNA5 WMMA types ----------------
typedef __attribute__((ext_vector_type(16))) __bf16 v16bf;
typedef __attribute__((ext_vector_type(8)))  float  v8f;
typedef unsigned int   u32x4 __attribute__((ext_vector_type(4)));
typedef unsigned short u16x4 __attribute__((ext_vector_type(4)));

union Frag { v16bf v; u32x4 q[2]; };

__device__ __forceinline__ unsigned short f2bf(float f) {
  unsigned int u = __float_as_uint(f);
  u += 0x7FFFu + ((u >> 16) & 1u);          // round-to-nearest-even
  return (unsigned short)(u >> 16);
}
__device__ __forceinline__ float bf2f(unsigned short h) {
  return __uint_as_float(((unsigned int)h) << 16);
}

// 4 bilinear taps per (pixel, kernel-tap): xnhwc plane offset + weight
// (bilinear weight * modulation folded in). BRANCHLESS: invalid taps carry
// idx=0 (safe to load) and wgt=0 (kills the contribution) -> no exec-mask
// branches in the GEMM hot loop.
struct Tap4 { int idx[4]; float wgt[4]; };

// =================== pre-pack kernels ===================

// x [B][C][H][W] f32 -> xnhwc [B][H][W][C] bf16 (gathers become channel-contiguous)
__global__ void pack_x_nhwc(const float* __restrict__ x, unsigned short* __restrict__ xo) {
  size_t i = (size_t)blockIdx.x * 256 + threadIdx.x;   // over M_TOT*CIN
  int c = (int)(i & 2047);
  size_t rest = i >> 11;            // b*HW + pix
  int pix = (int)(rest & 1023);
  int b = (int)(rest >> 10);
  xo[i] = f2bf(x[((size_t)(b * CIN + c)) * HW + pix]);
}

// plain f32 -> bf16 (row-major preserved); used for wfuse [256][1280]
__global__ void pack_bf16(const float* __restrict__ s, unsigned short* __restrict__ d, int n) {
  int i = blockIdx.x * 256 + threadIdx.x;
  if (i < n) d[i] = f2bf(s[i]);
}

// main deform weights w[o][c][kk] -> wbT[o][n*CIN + c] bf16 (K-major rows)
__global__ void pack_wbT(const float* __restrict__ w, unsigned short* __restrict__ dst, int ntaps) {
  size_t total = (size_t)COUT * ntaps * CIN;
  size_t i = (size_t)blockIdx.x * 256 + threadIdx.x;
  if (i >= total) return;
  int c  = (int)(i & 2047);
  size_t on = i >> 11;
  int nn = (int)(on % ntaps);
  int o  = (int)(on / ntaps);
  dst[i] = f2bf(w[((size_t)o * CIN + c) * ntaps + nn]);
}

// offset+mod conv weights packed as one padded B matrix:
// rows j: [0,2n) = woff[j], [2n,3n) = wm[j-2n], [3n,64) = 0 ; layout dst[j][pos*CIN + c]
__global__ void pack_wbTo(const float* __restrict__ woff, const float* __restrict__ wm,
                          unsigned short* __restrict__ dst, int n) {
  size_t i = (size_t)blockIdx.x * 256 + threadIdx.x;   // over 64*9*CIN
  int c   = (int)(i & 2047);
  int pos = (int)((i >> 11) % 9);
  int j   = (int)(i / (9 * CIN));
  float v = 0.f;
  if (j < 2 * n)      v = woff[((size_t)j * CIN + c) * 9 + pos];
  else if (j < 3 * n) v = wm[((size_t)(j - 2 * n) * CIN + c) * 9 + pos];
  dst[i] = f2bf(v);
}

// =================== geometry kernels ===================

// identity im2col taps for the 3x3/pad1/dil1 offset+mod conv
__global__ void build_geom_im2col(Tap4* __restrict__ g) {
  int t = blockIdx.x * 256 + threadIdx.x;              // over M_TOT*9
  if (t >= M_TOT * 9) return;
  int m = t / 9, pos = t - m * 9;
  int b = m >> 10, p = m & 1023, h = p >> 5, w = p & 31;
  int yy = h + pos / 3 - 1, xx = w + pos % 3 - 1;
  bool ok = ((unsigned)yy < 32u) && ((unsigned)xx < 32u);
  Tap4 tp;
  tp.idx[0] = ok ? ((b * 32 + yy) * 32 + xx) * CIN : 0;
  tp.wgt[0] = ok ? 1.f : 0.f;
  tp.idx[1] = tp.idx[2] = tp.idx[3] = 0;
  tp.wgt[1] = tp.wgt[2] = tp.wgt[3] = 0.f;
  g[t] = tp;
}

// deformable taps: offbuf [M_TOT][64] rows = {off_y[0..n), off_x[0..n), mod_logit[0..n)}
__global__ void build_geom_deform(const float* __restrict__ offbuf, Tap4* __restrict__ g,
                                  int k, int dil, int pad, int ntaps) {
  int t = blockIdx.x * 256 + threadIdx.x;              // over M_TOT*ntaps
  if (t >= M_TOT * ntaps) return;
  int m = t / ntaps, n = t - m * ntaps;
  int b = m >> 10, p = m & 1023, h = p >> 5, w = p & 31;
  int ky = n / k, kx = n - ky * k;
  const float* om = offbuf + (size_t)m * 64;
  float offy = om[n];
  float offx = om[ntaps + n];
  float mod  = 1.f / (1.f + __expf(-om[2 * ntaps + n]));
  int hpi = 32 + 2 * pad;
  float py = (float)(h + ky * dil) + offy;
  float px = (float)(w + kx * dil) + offx;
  py = fminf(fmaxf(py, 0.f), (float)(hpi - 1));
  px = fminf(fmaxf(px, 0.f), (float)(hpi - 1));
  float y0f = floorf(py), x0f = floorf(px);
  float wy = py - y0f, wx = px - x0f;
  int y0 = min(max((int)y0f, 0), hpi - 1);
  int x0 = min(max((int)x0f, 0), hpi - 1);
  int y1 = min(y0 + 1, hpi - 1);
  int x1 = min(x0 + 1, hpi - 1);
  float ws4[4] = { (1.f - wy) * (1.f - wx) * mod, (1.f - wy) * wx * mod,
                   wy * (1.f - wx) * mod,         wy * wx * mod };
  int ys[2] = { y0, y1 }, xs[2] = { x0, x1 };
  Tap4 tp;
  #pragma unroll
  for (int j = 0; j < 4; ++j) {
    int yy = ys[j >> 1] - pad, xx = xs[j & 1] - pad;
    bool ok = ((unsigned)yy < 32u) && ((unsigned)xx < 32u);
    tp.idx[j] = ok ? ((b * 32 + yy) * 32 + xx) * CIN : 0;
    tp.wgt[j] = ok ? ws4[j] : 0.f;
  }
  g[t] = tp;
}

// =================== WMMA GEMM kernels ===================
//
// deform_gemm<NT>: out[m][o] = sum_{n,c} (sum_taps wgt * x[b,c,tap]) * wbT[o][n*CIN+c]
// block = 128 threads (4 waves). Block tile: 16 (M) x (64*NT) (N); each wave owns
// NT 16x16 subtiles, so one LDS A fragment feeds NT v_wmma per K-step.
// A tile (16x32 bf16) built cooperatively, DOUBLE-BUFFERED in LDS -> a single
// __syncthreads per K-step. LDS rows padded to 40 ushorts (80B): 16B-aligned
// ds_load_b128 + bank spread.
template <int NT>
__global__ __launch_bounds__(128) void deform_gemm(
    const unsigned short* __restrict__ xnhwc,
    const unsigned short* __restrict__ wbT,   // [Nrows][ntaps*CIN] bf16, K-major rows
    const Tap4* __restrict__ geom,            // [M_TOT][ntaps]
    float* __restrict__ out,                  // [M_TOT][Nw]
    int ntaps, int Nw) {
  __shared__ __align__(16) unsigned short Alds[2][16][40];
  const int tid   = threadIdx.x;
  const int lane  = tid & 31;
  const int wave  = tid >> 5;
  const int mbase = blockIdx.x * 16;
  const int obase = (blockIdx.y * 4 + wave) * (16 * NT);
  const int Ktot  = ntaps * CIN;

  const int am  = tid >> 3;                 // A-build: row 0..15
  const int akg = tid & 7;                  // A-build: k-group, 4 channels each
  const int lm  = lane & 15;
  const int kb  = (lane < 16) ? 0 : 8;      // ISA 16-bit A/B fragment K layout

  const unsigned short* brow = wbT + (size_t)(obase + lm) * Ktot + kb;

  v8f acc[NT];
  #pragma unroll
  for (int t = 0; t < NT; ++t) acc[t] = {};

  int pp = 0;
  for (int n = 0; n < ntaps; ++n) {
    Tap4 t4 = geom[(size_t)(mbase + am) * ntaps + n];  // shared by 32 K-chunk channels
    for (int c = 0; c < CIN; c += 32) {
      // ---- build 16x32 bf16 A tile: branchless, 4 taps x 4 channels/thread ----
      const int c0 = c + akg * 4;
      float v0 = 0.f, v1 = 0.f, v2 = 0.f, v3 = 0.f;
      #pragma unroll
      for (int j = 0; j < 4; ++j) {
        u16x4 xv = *(const u16x4*)(xnhwc + (size_t)t4.idx[j] + c0);  // 8B coalesced
        float wg = t4.wgt[j];
        v0 += wg * bf2f(xv.x);
        v1 += wg * bf2f(xv.y);
        v2 += wg * bf2f(xv.z);
        v3 += wg * bf2f(xv.w);
      }
      u16x4 st = { f2bf(v0), f2bf(v1), f2bf(v2), f2bf(v3) };
      *(u16x4*)&Alds[pp][am][akg * 4] = st;
      __syncthreads();

      // ---- fragments per ISA layout: lanes<16 K{0..7,16..23}, lanes>=16 K{8..15,24..31}
      Frag a;
      a.q[0] = *(const u32x4*)&Alds[pp][lm][kb];
      a.q[1] = *(const u32x4*)&Alds[pp][lm][kb + 16];
      const size_t koff = (size_t)n * CIN + c;
      #pragma unroll
      for (int t = 0; t < NT; ++t) {
        Frag bfr;
        const unsigned short* bptr = brow + (size_t)t * 16 * Ktot + koff;
        bfr.q[0] = *(const u32x4*)bptr;
        bfr.q[1] = *(const u32x4*)(bptr + 16);
        __builtin_prefetch(bptr + 32, 0, 1);
        acc[t] = __builtin_amdgcn_wmma_f32_16x16x32_bf16(
            false, a.v, false, bfr.v, (short)0, acc[t], false, false);
      }
      pp ^= 1;   // double buffer: next store goes to the other A tile
    }
  }
  // C/D layout: VGPR r -> lanes 0..15: (M=r, N=lane) ; lanes 16..31: (M=8+r, N=lane-16)
  const int mr = mbase + ((lane < 16) ? 0 : 8);
  #pragma unroll
  for (int t = 0; t < NT; ++t) {
    const int co = obase + t * 16 + lm;
    #pragma unroll
    for (int r = 0; r < 8; ++r)
      out[(size_t)(mr + r) * Nw + co] = acc[t][r];
  }
}

// fuse GEMM: out[m][o] = sum_k cat[m][k] * wfuseT[o][k]; pure register WMMA loop,
// each wave owns two 16x16 subtiles (A fragment reused twice).
__global__ __launch_bounds__(128) void fuse_gemm(
    const unsigned short* __restrict__ A,    // [M_TOT][CCAT] bf16
    const unsigned short* __restrict__ Bw,   // [COUT][CCAT] bf16
    float* __restrict__ out) {               // [M_TOT][COUT]
  const int lane  = threadIdx.x & 31;
  const int wave  = threadIdx.x >> 5;
  const int mbase = blockIdx.x * 16;
  const int obase = (blockIdx.y * 4 + wave) * 32;
  const int lm = lane & 15;
  const int kb = (lane < 16) ? 0 : 8;
  const unsigned short* arow  = A  + (size_t)(mbase + lm) * CCAT + kb;
  const unsigned short* brow0 = Bw + (size_t)(obase + lm) * CCAT + kb;
  const unsigned short* brow1 = brow0 + (size_t)16 * CCAT;
  v8f acc0 = {}, acc1 = {};
  for (int k = 0; k < CCAT; k += 32) {
    Frag a, b0, b1;
    a.q[0]  = *(const u32x4*)(arow + k);
    a.q[1]  = *(const u32x4*)(arow + k + 16);
    b0.q[0] = *(const u32x4*)(brow0 + k);
    b0.q[1] = *(const u32x4*)(brow0 + k + 16);
    b1.q[0] = *(const u32x4*)(brow1 + k);
    b1.q[1] = *(const u32x4*)(brow1 + k + 16);
    acc0 = __builtin_amdgcn_wmma_f32_16x16x32_bf16(
        false, a.v, false, b0.v, (short)0, acc0, false, false);
    acc1 = __builtin_amdgcn_wmma_f32_16x16x32_bf16(
        false, a.v, false, b1.v, (short)0, acc1, false, false);
  }
  const int mr = mbase + ((lane < 16) ? 0 : 8);
  #pragma unroll
  for (int r = 0; r < 8; ++r) {
    out[(size_t)(mr + r) * COUT + obase + lm]      = acc0[r];
    out[(size_t)(mr + r) * COUT + obase + 16 + lm] = acc1[r];
  }
}

// =================== BN / misc kernels ===================

// per-channel batch-stat BN folded to scale/shift; s is [M_TOT][COUT]
__global__ void bn_stats(const float* __restrict__ s, const float* __restrict__ g,
                         const float* __restrict__ bta,
                         float* __restrict__ scale, float* __restrict__ shift) {
  __shared__ float ssum[128], ssq[128];
  const int o = blockIdx.x;        // 256 blocks
  const int t = threadIdx.x;       // 128 threads
  float sum = 0.f, sq = 0.f;
  for (int m = t; m < M_TOT; m += 128) {
    float v = s[(size_t)m * COUT + o];
    sum += v; sq += v * v;
  }
  ssum[t] = sum; ssq[t] = sq;
  __syncthreads();
  for (int st = 64; st > 0; st >>= 1) {
    if (t < st) { ssum[t] += ssum[t + st]; ssq[t] += ssq[t + st]; }
    __syncthreads();
  }
  if (t == 0) {
    float mu  = ssum[0] * (1.f / M_TOT);
    float var = ssq[0] * (1.f / M_TOT) - mu * mu;
    float sc  = g[o] * rsqrtf(var + 1e-5f);
    scale[o] = sc;
    shift[o] = bta[o] - mu * sc;
  }
}

__global__ void bn_relu_to_cat(const float* __restrict__ s, const float* __restrict__ scale,
                               const float* __restrict__ shift,
                               unsigned short* __restrict__ cat, int coloff) {
  int i = blockIdx.x * 256 + threadIdx.x;   // over M_TOT*COUT
  int m = i >> 8, o = i & 255;
  float v = fmaxf(s[(size_t)i] * scale[o] + shift[o], 0.f);
  cat[(size_t)m * CCAT + coloff + o] = f2bf(v);
}

// ----- pooled branch -----
__global__ void gap_kernel(const float* __restrict__ x, float* __restrict__ gap) {
  int i = blockIdx.x * 256 + threadIdx.x;   // over B_*CIN
  const float* px = x + (size_t)i * HW;
  float s = 0.f;
  for (int p = 0; p < HW; ++p) s += px[p];
  gap[i] = s * (1.f / HW);
}

__global__ void pool_fc(const float* __restrict__ gap, const float* __restrict__ wpool,
                        float* __restrict__ pv) {
  int i = blockIdx.x * 256 + threadIdx.x;   // over B_*COUT
  int b = i >> 8, o = i & 255;
  const float* g  = gap + (size_t)b * CIN;
  const float* wr = wpool + (size_t)o * CIN;
  float s = 0.f;
  for (int c = 0; c < CIN; ++c) s += g[c] * wr[c];
  pv[i] = s;
}

__global__ void pool_bn(const float* __restrict__ pv, const float* __restrict__ gp,
                        const float* __restrict__ bp, float* __restrict__ pact) {
  int o = threadIdx.x;                      // 256
  float v0 = pv[o], v1 = pv[COUT + o];
  float mu  = 0.5f * (v0 + v1);
  float var = 0.5f * ((v0 - mu) * (v0 - mu) + (v1 - mu) * (v1 - mu));
  float sc  = gp[o] * rsqrtf(var + 1e-5f);
  pact[o]        = fmaxf((v0 - mu) * sc + bp[o], 0.f);
  pact[COUT + o] = fmaxf((v1 - mu) * sc + bp[o], 0.f);
}

__global__ void pool_to_cat(const float* __restrict__ pact, unsigned short* __restrict__ cat) {
  int i = blockIdx.x * 256 + threadIdx.x;   // over M_TOT*COUT
  int o = i & 255, m = i >> 8, b = m >> 10;
  cat[(size_t)m * CCAT + 1024 + o] = f2bf(pact[b * COUT + o]);
}

// final BN+ReLU, write NCHW output
__global__ void final_out(const float* __restrict__ s2, const float* __restrict__ scale,
                          const float* __restrict__ shift, float* __restrict__ out) {
  int i = blockIdx.x * 256 + threadIdx.x;   // over B_*COUT*HW
  int p = i & 1023;
  int o = (i >> 10) & 255;
  int b = i >> 18;
  float v = s2[(size_t)(b * HW + p) * COUT + o] * scale[o] + shift[o];
  out[i] = fmaxf(v, 0.f);
}

// =================== host orchestration ===================
extern "C" void kernel_launch(void* const* d_in, const int* in_sizes, int n_in,
                              void* d_out, int out_size, void* d_ws, size_t ws_size,
                              hipStream_t stream) {
  (void)in_sizes; (void)n_in; (void)out_size; (void)ws_size;
  const float* x = (const float*)d_in[0];
  struct BrIn { const float *woff, *wm, *w, *g, *b; int k, dil, pad; };
  BrIn br[4] = {
    { (const float*)d_in[1],  (const float*)d_in[2],  (const float*)d_in[3],
      (const float*)d_in[4],  (const float*)d_in[5],  1, 1,  0 },
    { (const float*)d_in[6],  (const float*)d_in[7],  (const float*)d_in[8],
      (const float*)d_in[9],  (const float*)d_in[10], 3, 6,  6 },
    { (const float*)d_in[11], (const float*)d_in[12], (const float*)d_in[13],
      (const float*)d_in[14], (const float*)d_in[15], 3, 12, 12 },
    { (const float*)d_in[16], (const float*)d_in[17], (const float*)d_in[18],
      (const float*)d_in[19], (const float*)d_in[20], 3, 18, 18 },
  };
  const float* wpool = (const float*)d_in[21];
  const float* gp    = (const float*)d_in[22];
  const float* bp    = (const float*)d_in[23];
  const float* wfuse = (const float*)d_in[24];
  const float* gf    = (const float*)d_in[25];
  const float* bfv   = (const float*)d_in[26];

  // ---- workspace bump allocator (~38 MB total) ----
  char* wsp = (char*)d_ws;
  auto take = [&](size_t bytes) -> void* {
    void* p = (void*)wsp;
    wsp += (bytes + 255) & ~(size_t)255;
    return p;
  };
  unsigned short* xnhwc  = (unsigned short*)take((size_t)M_TOT * CIN * 2);      // 8.39 MB
  unsigned short* wfuseT = (unsigned short*)take((size_t)COUT * CCAT * 2);      // 0.66 MB
  unsigned short* wbTo   = (unsigned short*)take((size_t)64 * 9 * CIN * 2);     // 2.36 MB
  unsigned short* wbT    = (unsigned short*)take((size_t)COUT * 9 * CIN * 2);   // 9.44 MB
  float*          offbuf = (float*)take((size_t)M_TOT * 64 * 4);                // 0.52 MB
  Tap4*           geomI  = (Tap4*)take((size_t)M_TOT * 9 * sizeof(Tap4));       // 0.59 MB
  Tap4*           geomD  = (Tap4*)take((size_t)M_TOT * 9 * sizeof(Tap4));       // 0.59 MB
  float*          stage1 = (float*)take((size_t)4 * M_TOT * COUT * 4);          // 8.39 MB
  unsigned short* catb   = (unsigned short*)take((size_t)M_TOT * CCAT * 2);     // 5.24 MB
  float*          stage2 = (float*)take((size_t)M_TOT * COUT * 4);              // 2.10 MB
  float*          gapb   = (float*)take((size_t)B_ * CIN * 4);
  float*          pv     = (float*)take((size_t)B_ * COUT * 4);
  float*          pact   = (float*)take((size_t)B_ * COUT * 4);
  float*          scl    = (float*)take((size_t)COUT * 4);
  float*          shf    = (float*)take((size_t)COUT * 4);

  // ---- one-time packs ----
  pack_x_nhwc<<<(M_TOT * CIN) / 256, 256, 0, stream>>>(x, xnhwc);
  pack_bf16<<<(COUT * CCAT + 255) / 256, 256, 0, stream>>>(wfuse, wfuseT, COUT * CCAT);
  build_geom_im2col<<<(M_TOT * 9 + 255) / 256, 256, 0, stream>>>(geomI);

  // ---- pooled branch ----
  gap_kernel<<<(B_ * CIN) / 256, 256, 0, stream>>>(x, gapb);
  pool_fc<<<(B_ * COUT) / 256, 256, 0, stream>>>(gapb, wpool, pv);
  pool_bn<<<1, 256, 0, stream>>>(pv, gp, bp, pact);
  pool_to_cat<<<(M_TOT * COUT) / 256, 256, 0, stream>>>(pact, catb);

  // ---- deformable branches ----
  for (int i = 0; i < 4; ++i) {
    const int k = br[i].k, n = k * k;
    // offset + modulation conv as WMMA GEMM (identity im2col taps, N padded to 64)
    pack_wbTo<<<(64 * 9 * CIN) / 256, 256, 0, stream>>>(br[i].woff, br[i].wm, wbTo, n);
    deform_gemm<1><<<dim3(128, 1), 128, 0, stream>>>(xnhwc, wbTo, geomI, offbuf, 9, 64);
    // taps from offsets/modulation
    build_geom_deform<<<(M_TOT * n + 255) / 256, 256, 0, stream>>>(
        offbuf, geomD, k, br[i].dil, br[i].pad, n);
    // main deformable GEMM: each wave does 2 N-subtiles (2 wmma per A tile)
    pack_wbT<<<((size_t)COUT * n * CIN + 255) / 256, 256, 0, stream>>>(br[i].w, wbT, n);
    deform_gemm<2><<<dim3(128, 2), 128, 0, stream>>>(
        xnhwc, wbT, geomD, stage1 + (size_t)i * M_TOT * COUT, n, COUT);
  }

  // ---- BN + ReLU into concat buffer ----
  for (int i = 0; i < 4; ++i) {
    const float* s1 = stage1 + (size_t)i * M_TOT * COUT;
    bn_stats<<<COUT, 128, 0, stream>>>(s1, br[i].g, br[i].b, scl, shf);
    bn_relu_to_cat<<<(M_TOT * COUT) / 256, 256, 0, stream>>>(s1, scl, shf, catb, i * COUT);
  }

  // ---- fuse 1x1 conv (WMMA), final BN + ReLU ----
  fuse_gemm<<<dim3(128, 2), 128, 0, stream>>>(catb, wfuseT, stage2);
  bn_stats<<<COUT, 128, 0, stream>>>(stage2, gf, bfv, scl, shf);
  final_out<<<(B_ * COUT * HW) / 256, 256, 0, stream>>>(stage2, scl, shf, (float*)d_out);
}
